// VarianceAdaptor_11639361372306
// MI455X (gfx1250) — matrix operand order repaired and checked by reading, not compiled, not run on hardware
//
#include <hip/hip_runtime.h>
#include <hip/hip_bf16.h>

#define BB 32
#define LL 256
#define HH 256
#define FF 256
#define KT 3
#define NB 256
#define ML 2048
#define EPSV 1e-5f

typedef __bf16 bf16;
typedef __attribute__((ext_vector_type(16))) __bf16 v16bf;
typedef __attribute__((ext_vector_type(8)))  float  v8f;

union V16U { uint4 u[2]; v16bf v; };

// ---------------------------------------------------------------------------
// Weight transpose + f32->bf16: src (i,k,c,f) row-major -> dst (i,k,f,c)
// ---------------------------------------------------------------------------
__global__ __launch_bounds__(256)
void transpose_w(const float* __restrict__ src, bf16* __restrict__ dst) {
  size_t id = (size_t)blockIdx.x * 256 + threadIdx.x;   // 3*3*256*256 total
  int f  = (int)(id & 255);
  int c  = (int)((id >> 8) & 255);
  int ik = (int)(id >> 16);                              // i*3 + k
  float v = src[id];
  dst[(((size_t)ik * FF + f) * HH) + c] = (bf16)v;
}

// ---------------------------------------------------------------------------
// Bucketize + embedding add; emit bf16 copies of x, x+pemb, and f32 x3
// ---------------------------------------------------------------------------
__global__ __launch_bounds__(256)
void embed_add(const float* __restrict__ x,
               const float* __restrict__ pt, const float* __restrict__ et,
               const float* __restrict__ pbins, const float* __restrict__ ebins,
               const float* __restrict__ pemb, const float* __restrict__ eemb,
               bf16* __restrict__ x_bf, bf16* __restrict__ x2_bf,
               float* __restrict__ x3) {
  int bl = blockIdx.x;            // b*L + l
  int h  = threadIdx.x;
  __shared__ int s_pi, s_ei;
  if (h == 0) {
    float pv = pt[bl], ev = et[bl];
    int lo = 0, hi = NB - 1;      // bins length = NBINS-1 = 255
    while (lo < hi) { int m = (lo + hi) >> 1; if (pbins[m] < pv) lo = m + 1; else hi = m; }
    s_pi = lo;
    lo = 0; hi = NB - 1;
    while (lo < hi) { int m = (lo + hi) >> 1; if (ebins[m] < ev) lo = m + 1; else hi = m; }
    s_ei = lo;
  }
  __syncthreads();
  size_t base = (size_t)bl * HH + h;
  float v  = x[base];
  float v2 = v  + pemb[(size_t)s_pi * HH + h];
  float v3 = v2 + eemb[(size_t)s_ei * HH + h];
  x_bf[base]  = (bf16)v;
  x2_bf[base] = (bf16)v2;
  x3[base]    = v3;
}

// ---------------------------------------------------------------------------
// Fused: Conv1d(K=3) via WMMA bf16 -> bias -> ReLU -> LayerNorm ->
//        (stage1) bf16 restore  OR  (stage2) Linear head + mask
// Block = 16 rows x 256 features, 8 waves, each wave owns 2 N-tiles of 16.
// A slab (18 rows, zero-padded at batch edges) staged in LDS once per block.
// ---------------------------------------------------------------------------
#define AROW 264   // padded LDS row stride in elements (528 B -> bank skew)

__global__ __launch_bounds__(256)
void predictor_stage(const bf16* __restrict__ in,      // [B*L][HH] bf16
                     const bf16* __restrict__ wb,      // [KT][FF][HH] bf16
                     const float* __restrict__ bias,
                     const float* __restrict__ gamma,
                     const float* __restrict__ beta,
                     bf16* __restrict__ out_bf,        // stage1 output (or null)
                     const float* __restrict__ wl,     // stage2 head (or null)
                     const float* __restrict__ blp,
                     const unsigned char* __restrict__ mask,
                     float* __restrict__ pred_out) {
  alignas(16) __shared__ bf16 sA[18 * AROW];   // 18 input rows, zero-padded
  __shared__ float tile[16][264];
  __shared__ float ps[16][16];
  __shared__ float ps2[16][16];
  __shared__ float smean[16], srstd[16];

  const int tid  = threadIdx.x;
  const int lane = tid & 31;
  const int wave = tid >> 5;
  const int half = lane >> 4;          // 0/1 lane half
  const int l16  = lane & 15;

  const int rg0 = blockIdx.x * 16;     // global row base (b*L + l), 16-aligned
  const int b   = rg0 / LL;
  const int l0  = rg0 % LL;

  // -------- stage the A slab: rows l0-1 .. l0+16, zero OOB rows ------------
  // 18 rows x 32 uint4 (16B) chunks = 576 chunks, 256 threads
  for (int ci = tid; ci < 18 * 32; ci += 256) {
    int r   = ci >> 5;
    int c16 = ci & 31;
    int lp  = l0 + r - 1;
    uint4 val = make_uint4(0u, 0u, 0u, 0u);
    if (lp >= 0 && lp < LL)
      val = *reinterpret_cast<const uint4*>(in + ((size_t)(b * LL + lp) * HH) + c16 * 8);
    *reinterpret_cast<uint4*>(&sA[r * AROW + c16 * 8]) = val;
  }
  __syncthreads();

  v8f acc0 = {}; v8f acc1 = {};
  const int f0  = wave * 32;
  const int fc0 = f0 + l16;            // B-tile columns for this lane
  const int fc1 = fc0 + 16;
  const int kb  = half * 16;           // B fragment K-base within 32-chunk
  const int ah  = half * 16;           // A fragment K-base within 32-chunk

#pragma unroll
  for (int k = 0; k < KT; ++k) {
    const bf16* arow   = &sA[(l16 + k) * AROW];
    const bf16* bbase0 = wb + (((size_t)k * FF + fc0) * HH);
    const bf16* bbase1 = wb + (((size_t)k * FF + fc1) * HH);
    if (k + 1 < KT)
      __builtin_prefetch(wb + (((size_t)(k + 1) * FF + fc0) * HH), 0, 0);
#pragma unroll
    for (int c0 = 0; c0 < HH; c0 += 32) {
      V16U a;
      a.u[0] = *reinterpret_cast<const uint4*>(arow + c0 + ah);
      a.u[1] = *reinterpret_cast<const uint4*>(arow + c0 + ah + 8);
      V16U bm0, bm1;
      const uint4* bp0 = reinterpret_cast<const uint4*>(bbase0 + c0 + kb);
      bm0.u[0] = bp0[0]; bm0.u[1] = bp0[1];
      const uint4* bp1 = reinterpret_cast<const uint4*>(bbase1 + c0 + kb);
      bm1.u[0] = bp1[0]; bm1.u[1] = bp1[1];
      acc0 = __builtin_amdgcn_wmma_f32_16x16x32_bf16(false, a.v, false, bm0.v,
                                                     (short)0, acc0, false, false);
      acc1 = __builtin_amdgcn_wmma_f32_16x16x32_bf16(false, a.v, false, bm1.v,
                                                     (short)0, acc1, false, false);
    }
  }
  __syncthreads();   // sA no longer needed; tile phase begins

  // bias + ReLU into LDS tile (C/D layout: VGPR r -> M=r+half*8, N=lane&15)
  {
    int n0 = f0 + l16;
    float bia0 = bias[n0], bia1 = bias[n0 + 16];
#pragma unroll
    for (int r = 0; r < 8; ++r) {
      int m = r + half * 8;
      float v0 = acc0[r] + bia0; v0 = v0 > 0.f ? v0 : 0.f;
      float v1 = acc1[r] + bia1; v1 = v1 > 0.f ? v1 : 0.f;
      tile[m][n0]      = v0;
      tile[m][n0 + 16] = v1;
    }
  }
  __syncthreads();

  // per-row mean / var partials: 16 threads per row, 16 elems each
  {
    int row = tid >> 4, seg = tid & 15;
    float s = 0.f, s2 = 0.f;
#pragma unroll
    for (int j = 0; j < 16; ++j) {
      float v = tile[row][seg * 16 + j];
      s += v; s2 += v * v;
    }
    ps[row][seg] = s; ps2[row][seg] = s2;
  }
  __syncthreads();
  if (tid < 16) {
    float s = 0.f, s2 = 0.f;
#pragma unroll
    for (int j = 0; j < 16; ++j) { s += ps[tid][j]; s2 += ps2[tid][j]; }
    float mean = s * (1.f / FF);
    float var  = s2 * (1.f / FF) - mean * mean;
    smean[tid] = mean;
    srstd[tid] = rsqrtf(var + EPSV);
  }
  __syncthreads();

  // normalize + emit
  {
    int row = tid >> 4, seg = tid & 15;
    float mean = smean[row], rstd = srstd[row];
    float dot = 0.f;
#pragma unroll
    for (int j = 0; j < 16; ++j) {
      int f = seg * 16 + j;
      float v = (tile[row][f] - mean) * rstd * gamma[f] + beta[f];
      if (out_bf)   out_bf[((size_t)(rg0 + row)) * FF + f] = (bf16)v;
      if (pred_out) dot += v * wl[f];
    }
    if (pred_out) ps[row][seg] = dot;
  }
  if (pred_out) {
    __syncthreads();
    if (tid < 16) {
      float s = 0.f;
#pragma unroll
      for (int j = 0; j < 16; ++j) s += ps[tid][j];
      s += blp[0];
      int rg = rg0 + tid;
      pred_out[rg] = mask[rg] ? 0.f : s;
    }
  }
}

// ---------------------------------------------------------------------------
// Per-batch inclusive scan of duration; emit cum + mel_len (as float)
// ---------------------------------------------------------------------------
__global__ __launch_bounds__(256)
void cumsum_k(const int* __restrict__ dur, int* __restrict__ cum,
              float* __restrict__ mlen) {
  __shared__ int s[256];
  int bidx = blockIdx.x, t = threadIdx.x;
  s[t] = dur[bidx * LL + t];
  __syncthreads();
  for (int off = 1; off < 256; off <<= 1) {
    int v = (t >= off) ? s[t - off] : 0;
    __syncthreads();
    s[t] += v;
    __syncthreads();
  }
  cum[bidx * LL + t] = s[t];
  if (t == 255) mlen[bidx] = (float)s[255];
}

// ---------------------------------------------------------------------------
// Length regulation: one block per (b, t) output frame; 256 feature threads
// ---------------------------------------------------------------------------
__global__ __launch_bounds__(256)
void expand_k(const float* __restrict__ x3, const int* __restrict__ cum,
              float* __restrict__ out, float* __restrict__ mmask) {
  int bidx = blockIdx.x >> 11;           // / 2048
  int t    = blockIdx.x & (ML - 1);
  const int* c = cum + bidx * LL;
  __shared__ int s_idx, s_valid;
  if (threadIdx.x == 0) {
    int total = c[LL - 1];
    int lo = 0, hi = LL;                 // searchsorted right (upper_bound)
    while (lo < hi) { int m = (lo + hi) >> 1; if (c[m] <= t) lo = m + 1; else hi = m; }
    if (lo > LL - 1) lo = LL - 1;
    int valid = (t < total) ? 1 : 0;
    s_idx = lo; s_valid = valid;
    mmask[bidx * ML + t] = valid ? 0.f : 1.f;   // ~valid
  }
  __syncthreads();
  float v = s_valid ? x3[((size_t)(bidx * LL + s_idx)) * HH + threadIdx.x] : 0.f;
  out[((size_t)(bidx * ML + t)) * HH + threadIdx.x] = v;
}

// ---------------------------------------------------------------------------
extern "C" void kernel_launch(void* const* d_in, const int* in_sizes, int n_in,
                              void* d_out, int out_size, void* d_ws, size_t ws_size,
                              hipStream_t stream) {
  const float* x      = (const float*)d_in[0];
  const float* pt     = (const float*)d_in[1];
  const float* et     = (const float*)d_in[2];
  const float* w1     = (const float*)d_in[3];
  const float* b1     = (const float*)d_in[4];
  const float* g1     = (const float*)d_in[5];
  const float* be1    = (const float*)d_in[6];
  const float* w2     = (const float*)d_in[7];
  const float* b2     = (const float*)d_in[8];
  const float* g2     = (const float*)d_in[9];
  const float* be2    = (const float*)d_in[10];
  const float* wl     = (const float*)d_in[11];
  const float* bl     = (const float*)d_in[12];
  const float* pbins  = (const float*)d_in[13];
  const float* ebins  = (const float*)d_in[14];
  const float* pemb   = (const float*)d_in[15];
  const float* eemb   = (const float*)d_in[16];
  const unsigned char* smask = (const unsigned char*)d_in[17];
  const int*   dtarg  = (const int*)d_in[18];

  // workspace layout (bytes)
  char* ws = (char*)d_ws;
  bf16*  x_bf  = (bf16*)(ws);                                 // 4 MiB
  bf16*  x2_bf = (bf16*)(ws + 4194304);                       // 4 MiB
  float* x3    = (float*)(ws + 8388608);                      // 8 MiB
  bf16*  wb1   = (bf16*)(ws + 16777216);                      // 1.125 MiB
  bf16*  wb2   = (bf16*)(ws + 16777216 + 1179648);            // 1.125 MiB
  bf16*  hbuf  = (bf16*)(ws + 16777216 + 2 * 1179648);        // 4 MiB
  int*   cum   = (int*)(ws + 16777216 + 2 * 1179648 + 4194304);

  // output layout (flat, return order)
  float* out    = (float*)d_out;                    // B*ML*H
  float* dur_p  = out + (size_t)BB * ML * HH;       // B*L
  float* pit_p  = dur_p + BB * LL;
  float* ene_p  = pit_p + BB * LL;
  float* mlen   = ene_p + BB * LL;                  // B
  float* mmask  = mlen + BB;                        // B*ML

  transpose_w<<<2304, 256, 0, stream>>>(w1, wb1);
  transpose_w<<<2304, 256, 0, stream>>>(w2, wb2);
  embed_add<<<BB * LL, 256, 0, stream>>>(x, pt, et, pbins, ebins, pemb, eemb,
                                         x_bf, x2_bf, x3);

  const int nblk = (BB * LL) / 16;   // 512
  for (int i = 0; i < 3; ++i) {
    const bf16* in_act = (i == 2) ? x2_bf : x_bf;
    float* pred = (i == 0) ? dur_p : (i == 1) ? pit_p : ene_p;
    predictor_stage<<<nblk, 256, 0, stream>>>(
        in_act, wb1 + (size_t)i * KT * FF * HH,
        b1 + i * FF, g1 + i * FF, be1 + i * FF,
        hbuf, nullptr, nullptr, nullptr, nullptr);
    predictor_stage<<<nblk, 256, 0, stream>>>(
        hbuf, wb2 + (size_t)i * KT * FF * HH,
        b2 + i * FF, g2 + i * FF, be2 + i * FF,
        nullptr, wl + i * FF, bl + i, smask, pred);
  }

  cumsum_k<<<BB, 256, 0, stream>>>(dtarg, cum, mlen);
  expand_k<<<BB * ML, 256, 0, stream>>>(x3, cum, out, mmask);
}
